// LSTM_CRF_72971494359255
// MI455X (gfx1250) — matrix operand order repaired and checked by reading, not compiled, not run on hardware
//
#include <hip/hip_runtime.h>
#include <hip/hip_bf16.h>

// ---------------------------------------------------------------------------
// BiLSTM-CRF loss for MI455X (gfx1250, wave32, WMMA bf16 16x16x32).
// All matrix operands kept bf16 at rest (emb table pre-converted, h carried
// in bf16) so the hot loop is pure b128 loads + v_wmma, no converts.
// ---------------------------------------------------------------------------

typedef __attribute__((ext_vector_type(16))) __bf16 v16bf;
typedef __attribute__((ext_vector_type(8)))  __bf16 v8bf;
typedef __attribute__((ext_vector_type(8)))  float  v8f;

#define BB   128    // batch
#define SS   512    // sequence length
#define VV   50000  // vocab
#define EE   256    // embedding dim
#define HH   512    // hidden dim
#define G4   2048   // 4*H gate width
#define TT   32     // tags
#define KTOT 768    // E + H

__device__ __forceinline__ __bf16 f2bf(float x) { return (__bf16)x; }
__device__ __forceinline__ float  sigf(float x) { return 1.0f / (1.0f + __expf(-x)); }

// ------------------------------- utility -----------------------------------
__global__ void zero_kernel(float* __restrict__ p, int n) {
    int i = blockIdx.x * blockDim.x + threadIdx.x;
    if (i < n) p[i] = 0.0f;
}

// embB[i] = bf16(emb[i]), flat over V*E
__global__ void prep_emb_kernel(const float* __restrict__ emb,
                                __bf16* __restrict__ embB) {
    int i = blockIdx.x * blockDim.x + threadIdx.x;
    if (i < VV * EE) embB[i] = f2bf(emb[i]);
}

// Wcat[k][n] (bf16, row-major [768][2048]) = concat(W_ih^T, W_hh^T)
__global__ void prep_wcat_kernel(const float* __restrict__ w_ih,
                                 const float* __restrict__ w_hh,
                                 __bf16* __restrict__ Wcat) {
    int i = blockIdx.x * blockDim.x + threadIdx.x;
    if (i >= KTOT * G4) return;
    int k = i >> 11;        // 0..767
    int n = i & (G4 - 1);   // 0..2047
    float v = (k < EE) ? w_ih[n * EE + k] : w_hh[n * HH + (k - EE)];
    Wcat[i] = f2bf(v);
}

__global__ void prep_bsum_kernel(const float* __restrict__ b_ih,
                                 const float* __restrict__ b_hh,
                                 float* __restrict__ bsum) {
    int i = blockIdx.x * blockDim.x + threadIdx.x;
    if (i < G4) bsum[i] = b_ih[i] + b_hh[i];
}

// linT[k][n] (bf16, [512][32]) = lin_w[n][koff + k]   (lin_w is [32][1024])
__global__ void prep_lint_kernel(const float* __restrict__ lin_w,
                                 __bf16* __restrict__ linT, int koff) {
    int i = blockIdx.x * blockDim.x + threadIdx.x;
    if (i >= HH * TT) return;
    int k = i >> 5;        // 0..511
    int n = i & (TT - 1);  // 0..31
    linT[i] = f2bf(lin_w[n * (2 * HH) + koff + k]);
}

// --------------------------- fused LSTM step --------------------------------
// 64 blocks x 128 threads = 256 waves. Wave wid: row tile rt = wid>>5 (8 x 16
// batch rows), col tile nt = wid&31 (32 x 16 hidden units). Each wave computes
// the 4 gate tiles (i,f,g,o) sharing its (row,col) slice over K = E + H = 768,
// gathering bf16 x_t straight from the pre-converted embedding table (with
// per-batch reversal for the backward pass), applies activations + mask,
// writes bf16 h / fp32 c carries, and accumulates its partial contribution to
// the logits via two more WMMAs + scatter atomics.
__global__ __launch_bounds__(128)
void lstm_step_kernel(const int*    __restrict__ sents,
                      const int*    __restrict__ lengths,
                      const __bf16* __restrict__ embB,     // [V][256] bf16
                      const __bf16* __restrict__ Wcat,     // [768][2048] bf16
                      const float*  __restrict__ bsum,     // [2048]
                      const __bf16* __restrict__ linT,     // [512][32] bf16
                      const __bf16* __restrict__ h_in,     // [128][512] bf16
                      __bf16*       __restrict__ h_out,    // [128][512] bf16
                      float*        __restrict__ c,        // [128][512]
                      float*        __restrict__ logitsAcc,// [512][128][32]
                      int t, int backward)
{
    __shared__ __bf16 hstage[4][256];  // per-wave 16x16 bf16 masked-h tile

    const int lane     = threadIdx.x & 31;
    const int wslot    = threadIdx.x >> 5;
    const int wid      = blockIdx.x * 4 + wslot;
    const int rt       = wid >> 5;       // 0..7
    const int nt       = wid & 31;       // 0..31
    const int rowbase  = rt * 16;
    const int colbase  = nt * 16;
    const int laneHalf = lane >> 4;      // 0|1
    const int lcol     = lane & 15;

    // A-fragment row for this lane (lanes l and l+16 hold the same row l&15).
    const int arow = rowbase + lcol;
    const int alen = lengths[arow];
    const int sx   = backward ? ((t < alen) ? (alen - 1 - t) : t) : t;
    const __bf16* embrow = embB + (long)sents[arow * SS + sx] * EE;

    v8f acc0 = {}, acc1 = {}, acc2 = {}, acc3 = {};

    // ---- K over embedding input (k = 0..255): pure bf16 b128 loads ----
    for (int kb = 0; kb < EE; kb += 32) {
        const __bf16* src = embrow + kb + (laneHalf ? 8 : 0);
        v8bf alo = *(const v8bf*)(src);        // K+0..7   (this half)
        v8bf ahi = *(const v8bf*)(src + 16);   // K+16..23 (this half)
        v16bf a;
        #pragma unroll
        for (int e = 0; e < 8; ++e) { a[e] = alo[e]; a[8 + e] = ahi[e]; }
        v16bf b0, b1, b2, b3;
        #pragma unroll
        for (int e = 0; e < 16; ++e) {
            const int kk = kb + (e < 8 ? 8 * laneHalf + e : 16 + 8 * laneHalf + (e - 8));
            const __bf16* brow = Wcat + (long)kk * G4 + colbase + lcol;
            b0[e] = brow[0]; b1[e] = brow[HH]; b2[e] = brow[2 * HH]; b3[e] = brow[3 * HH];
        }
        acc0 = __builtin_amdgcn_wmma_f32_16x16x32_bf16(false, a, false, b0, (short)0, acc0, false, false);
        acc1 = __builtin_amdgcn_wmma_f32_16x16x32_bf16(false, a, false, b1, (short)0, acc1, false, false);
        acc2 = __builtin_amdgcn_wmma_f32_16x16x32_bf16(false, a, false, b2, (short)0, acc2, false, false);
        acc3 = __builtin_amdgcn_wmma_f32_16x16x32_bf16(false, a, false, b3, (short)0, acc3, false, false);
    }
    // ---- K over recurrent state (k = 256..767): bf16 h carry ----
    for (int kb = 0; kb < HH; kb += 32) {
        const __bf16* src = h_in + arow * HH + kb + (laneHalf ? 8 : 0);
        v8bf alo = *(const v8bf*)(src);
        v8bf ahi = *(const v8bf*)(src + 16);
        v16bf a;
        #pragma unroll
        for (int e = 0; e < 8; ++e) { a[e] = alo[e]; a[8 + e] = ahi[e]; }
        v16bf b0, b1, b2, b3;
        #pragma unroll
        for (int e = 0; e < 16; ++e) {
            const int kk = EE + kb + (e < 8 ? 8 * laneHalf + e : 16 + 8 * laneHalf + (e - 8));
            const __bf16* brow = Wcat + (long)kk * G4 + colbase + lcol;
            b0[e] = brow[0]; b1[e] = brow[HH]; b2[e] = brow[2 * HH]; b3[e] = brow[3 * HH];
        }
        acc0 = __builtin_amdgcn_wmma_f32_16x16x32_bf16(false, a, false, b0, (short)0, acc0, false, false);
        acc1 = __builtin_amdgcn_wmma_f32_16x16x32_bf16(false, a, false, b1, (short)0, acc1, false, false);
        acc2 = __builtin_amdgcn_wmma_f32_16x16x32_bf16(false, a, false, b2, (short)0, acc2, false, false);
        acc3 = __builtin_amdgcn_wmma_f32_16x16x32_bf16(false, a, false, b3, (short)0, acc3, false, false);
    }

    // ---- activations + state update (C layout: elem r -> row laneHalf*8+r, col lcol) ----
    const float bi = bsum[0 * HH + colbase + lcol];
    const float bf = bsum[1 * HH + colbase + lcol];
    const float bg = bsum[2 * HH + colbase + lcol];
    const float bo = bsum[3 * HH + colbase + lcol];
    const int rbase = rowbase + laneHalf * 8;
    const int col   = colbase + lcol;
    #pragma unroll
    for (int r = 0; r < 8; ++r) {
        const int row = rbase + r;
        const float si = sigf(acc0[r] + bi);
        const float sf = sigf(acc1[r] + bf);
        const float tg = tanhf(acc2[r] + bg);
        const float so = sigf(acc3[r] + bo);
        const float cp = c[row * HH + col];
        const float cn = sf * cp + si * tg;
        const float hn = so * tanhf(cn);
        const bool  mk = t < lengths[row];
        const float hp = (float)h_in[row * HH + col];
        h_out[row * HH + col] = f2bf(mk ? hn : hp);
        c[row * HH + col]     = mk ? cn : cp;
        hstage[wslot][(laneHalf * 8 + r) * 16 + lcol] = f2bf(mk ? hn : 0.0f);
    }
    __syncthreads();

    // ---- fused partial logits: (16x16 masked h tile) @ linT[16 x 32], K padded to 32 ----
    v16bf la;
    {
        const __bf16* src = &hstage[wslot][lcol * 16 + (laneHalf ? 8 : 0)];
        #pragma unroll
        for (int e = 0; e < 8; ++e) { la[e] = src[e]; la[8 + e] = f2bf(0.0f); }
    }
    v16bf lb0, lb1;
    {
        #pragma unroll
        for (int e = 0; e < 8; ++e) {
            const __bf16* lrow = linT + (colbase + laneHalf * 8 + e) * TT + lcol;
            lb0[e] = lrow[0]; lb1[e] = lrow[16];
            lb0[8 + e] = f2bf(0.0f); lb1[8 + e] = f2bf(0.0f);
        }
    }
    v8f p0 = {}, p1 = {};
    p0 = __builtin_amdgcn_wmma_f32_16x16x32_bf16(false, la, false, lb0, (short)0, p0, false, false);
    p1 = __builtin_amdgcn_wmma_f32_16x16x32_bf16(false, la, false, lb1, (short)0, p1, false, false);

    #pragma unroll
    for (int r = 0; r < 8; ++r) {
        const int row  = rbase + r;
        const int rlen = lengths[row];
        // backward h(rev,t) lands at sequence position rev(b,t); forward at t.
        const int sDst = backward ? ((t < rlen) ? (rlen - 1 - t) : t) : t;
        float* dst = logitsAcc + ((size_t)sDst * BB + row) * TT;
        atomicAdd(dst + lcol,      p0[r]);
        atomicAdd(dst + 16 + lcol, p1[r]);
    }
}

// ------------------------------ CRF NLL -------------------------------------
__device__ __forceinline__ float getE(const float* la, const float* lb,
                                      int t, int b, int tag) {
    return la[((size_t)t * BB + b) * TT + tag] + lb[tag];
}

// One wave32 per batch row; lane j owns tag state j.
__global__ __launch_bounds__(32)
void crf_kernel(const float* __restrict__ logitsAcc,
                const float* __restrict__ lin_b,
                const int*   __restrict__ tags,
                const int*   __restrict__ lengths,
                const float* __restrict__ startT,
                const float* __restrict__ endT,
                const float* __restrict__ trans,
                float*       __restrict__ out)
{
    const int b = blockIdx.x;
    const int j = threadIdx.x;
    const int len = lengths[b];

    float tcol[TT];                       // transition column j in registers
    #pragma unroll
    for (int i = 0; i < TT; ++i) tcol[i] = trans[i * TT + j];

    float alpha = startT[j] + getE(logitsAcc, lin_b, 0, b, j);
    float score = 0.0f; int prev = 0;
    if (j == 0) {
        int t0 = tags[b * SS];
        score = startT[t0] + getE(logitsAcc, lin_b, 0, b, t0);
        prev = t0;
    }

    for (int t = 1; t < SS; ++t) {
        float ej = getE(logitsAcc, lin_b, t, b, j);
        float m = -3.4e38f;
        for (int i = 0; i < TT; ++i) {
            float v = __shfl(alpha, i, 32) + tcol[i];
            m = fmaxf(m, v);
        }
        float s = 0.0f;
        for (int i = 0; i < TT; ++i) {
            float v = __shfl(alpha, i, 32) + tcol[i];
            s += __expf(v - m);
        }
        float nxt = m + __logf(s) + ej;
        if (t < len) alpha = nxt;                      // len uniform per wave
        if (j == 0 && t < len) {
            int tt = tags[b * SS + t];
            score += trans[prev * TT + tt] + getE(logitsAcc, lin_b, t, b, tt);
            prev = tt;
        }
    }

    float v = alpha + endT[j];
    float m = v;
    for (int off = 16; off > 0; off >>= 1) m = fmaxf(m, __shfl_xor(m, off, 32));
    float s = __expf(v - m);
    for (int off = 16; off > 0; off >>= 1) s += __shfl_xor(s, off, 32);
    if (j == 0) {
        float den = m + __logf(s);
        float num = score + endT[prev];
        atomicAdd(out, den - num);
    }
}

// ------------------------------- launcher -----------------------------------
extern "C" void kernel_launch(void* const* d_in, const int* in_sizes, int n_in,
                              void* d_out, int out_size, void* d_ws, size_t ws_size,
                              hipStream_t stream)
{
    const int*   sents   = (const int*)d_in[0];
    const int*   lengths = (const int*)d_in[1];
    const int*   tags    = (const int*)d_in[2];
    /* d_in[3] = mask; recomputed on device from lengths (bool layout varies) */
    const float* emb     = (const float*)d_in[4];
    const float* w_ih_f  = (const float*)d_in[5];
    const float* w_hh_f  = (const float*)d_in[6];
    const float* b_ih_f  = (const float*)d_in[7];
    const float* b_hh_f  = (const float*)d_in[8];
    const float* w_ih_b  = (const float*)d_in[9];
    const float* w_hh_b  = (const float*)d_in[10];
    const float* b_ih_b  = (const float*)d_in[11];
    const float* b_hh_b  = (const float*)d_in[12];
    const float* lin_w   = (const float*)d_in[13];
    const float* lin_b   = (const float*)d_in[14];
    const float* start_t = (const float*)d_in[15];
    const float* end_t   = (const float*)d_in[16];
    const float* trans   = (const float*)d_in[17];

    // ---- workspace carve-out (~42 MB) ----
    char* w = (char*)d_ws;
    __bf16* embB  = (__bf16*)w; w += (size_t)VV * EE * 2;       // 25.6 MB
    __bf16* WcatF = (__bf16*)w; w += (size_t)KTOT * G4 * 2;     // 3 MB
    __bf16* WcatB = (__bf16*)w; w += (size_t)KTOT * G4 * 2;     // 3 MB
    __bf16* linTF = (__bf16*)w; w += (size_t)HH * TT * 2;
    __bf16* linTB = (__bf16*)w; w += (size_t)HH * TT * 2;
    float*  bsumF = (float*)w;  w += (size_t)G4 * 4;
    float*  bsumB = (float*)w;  w += (size_t)G4 * 4;
    __bf16* hF0   = (__bf16*)w; w += (size_t)BB * HH * 2;
    __bf16* hF1   = (__bf16*)w; w += (size_t)BB * HH * 2;
    __bf16* hB0   = (__bf16*)w; w += (size_t)BB * HH * 2;
    __bf16* hB1   = (__bf16*)w; w += (size_t)BB * HH * 2;
    float*  cF    = (float*)w;  w += (size_t)BB * HH * 4;
    float*  cB    = (float*)w;  w += (size_t)BB * HH * 4;
    float*  logitsAcc = (float*)w; w += (size_t)SS * BB * TT * 4; // 8 MB

    // ---- re-initialize state every call (graph-replay safe) ----
    zero_kernel<<<1, 32, 0, stream>>>((float*)d_out, out_size);
    auto z = [&](float* p, int n) {
        zero_kernel<<<(n + 255) / 256, 256, 0, stream>>>(p, n);
    };
    // bf16 zeros share the 0x0000 bit pattern: zero as float over half count.
    z((float*)hF0, BB * HH / 2); z((float*)hF1, BB * HH / 2);
    z((float*)hB0, BB * HH / 2); z((float*)hB1, BB * HH / 2);
    z(cF, BB * HH);              z(cB, BB * HH);
    z(logitsAcc, SS * BB * TT);

    // ---- weight / table prep (transpose + bf16) ----
    prep_emb_kernel<<<(VV * EE + 255) / 256, 256, 0, stream>>>(emb, embB);
    prep_wcat_kernel<<<(KTOT * G4 + 255) / 256, 256, 0, stream>>>(w_ih_f, w_hh_f, WcatF);
    prep_wcat_kernel<<<(KTOT * G4 + 255) / 256, 256, 0, stream>>>(w_ih_b, w_hh_b, WcatB);
    prep_bsum_kernel<<<(G4 + 255) / 256, 256, 0, stream>>>(b_ih_f, b_hh_f, bsumF);
    prep_bsum_kernel<<<(G4 + 255) / 256, 256, 0, stream>>>(b_ih_b, b_hh_b, bsumB);
    prep_lint_kernel<<<(HH * TT + 255) / 256, 256, 0, stream>>>(lin_w, linTF, 0);
    prep_lint_kernel<<<(HH * TT + 255) / 256, 256, 0, stream>>>(lin_w, linTB, HH);

    // ---- sequential LSTM scan, both directions, h ping-pong ----
    for (int t = 0; t < SS; ++t) {
        __bf16* hinF  = (t & 1) ? hF1 : hF0;
        __bf16* houtF = (t & 1) ? hF0 : hF1;
        lstm_step_kernel<<<64, 128, 0, stream>>>(sents, lengths, embB, WcatF, bsumF,
                                                 linTF, hinF, houtF, cF, logitsAcc, t, 0);
        __bf16* hinB  = (t & 1) ? hB1 : hB0;
        __bf16* houtB = (t & 1) ? hB0 : hB1;
        lstm_step_kernel<<<64, 128, 0, stream>>>(sents, lengths, embB, WcatB, bsumB,
                                                 linTB, hinB, houtB, cB, logitsAcc, t, 1);
    }

    // ---- CRF negative log-likelihood ----
    crf_kernel<<<BB, 32, 0, stream>>>(logitsAcc, lin_b, tags, lengths,
                                      start_t, end_t, trans, (float*)d_out);
}
